// TransformerEncoderLayer_23252952940951
// MI455X (gfx1250) — matrix operand order repaired and checked by reading, not compiled, not run on hardware
//
#include <hip/hip_runtime.h>
#include <hip/hip_bf16.h>
#include <math.h>

// Problem dims (fixed by the reference)
#define BB  2
#define SS  2048
#define DDIM 1024
#define HH  16
#define HD  64
#define FF  4096
#define MM  (BB * SS)   // 4096 rows

typedef __attribute__((ext_vector_type(16))) __bf16 v16bf;
typedef __attribute__((ext_vector_type(8)))  float  v8f;

union BF16Frag { v16bf v; uint4 u[2]; };

__device__ inline v16bf load_frag2(const __bf16* p0, const __bf16* p1) {
  BF16Frag f;
  f.u[0] = *reinterpret_cast<const uint4*>(p0);
  f.u[1] = *reinterpret_cast<const uint4*>(p1);
  return f.v;
}

__device__ inline float gelu_exact(float x) {
  return 0.5f * x * (1.0f + erff(x * 0.70710678118654752f));
}

// ---------------------------------------------------------------------------
// elementwise f32 -> bf16 cast
// ---------------------------------------------------------------------------
__global__ void cast_bf16_kernel(const float* __restrict__ in,
                                 __bf16* __restrict__ out, int n) {
  int i = blockIdx.x * blockDim.x + threadIdx.x;
  if (i < n) out[i] = (__bf16)in[i];
}

// ---------------------------------------------------------------------------
// transpose + cast: in f32 [R, C] -> out bf16 [C, R]
// ---------------------------------------------------------------------------
__global__ void transpose_cast_kernel(const float* __restrict__ in,
                                      __bf16* __restrict__ out, int R, int C) {
  __shared__ float t[32][33];
  int c0 = blockIdx.x * 32, r0 = blockIdx.y * 32;
  int tx = threadIdx.x, ty = threadIdx.y;  // 32 x 8
  #pragma unroll
  for (int i = 0; i < 32; i += 8)
    t[ty + i][tx] = in[(size_t)(r0 + ty + i) * C + c0 + tx];
  __syncthreads();
  #pragma unroll
  for (int i = 0; i < 32; i += 8)
    out[(size_t)(c0 + ty + i) * R + r0 + tx] = (__bf16)t[tx][ty + i];
}

// ---------------------------------------------------------------------------
// GEMM: C[M,N] = A_bf16[M,K] * Bt_bf16[N,K]^T + bias (+res) (+gelu)
// 8 waves per WG in a 2x4 grid; wave tile 64x64 (4x4 WMMA accumulators);
// WG tile 128x256; K-step 32.  EPI: 0 = none, 1 = exact gelu.
// ---------------------------------------------------------------------------
template <int EPI>
__global__ __launch_bounds__(256)
void gemm_kernel(const __bf16* __restrict__ A, const __bf16* __restrict__ Bt,
                 const float* __restrict__ bias, const float* __restrict__ res,
                 float* __restrict__ Cf, __bf16* __restrict__ Cb,
                 int M, int N, int K) {
  const int lane = threadIdx.x & 31;
  const int wave = threadIdx.x >> 5;
  const int lh   = lane >> 4;     // lane half
  const int l16  = lane & 15;
  const int wm   = wave >> 2;     // 0..1
  const int wn   = wave & 3;      // 0..3
  const int m0   = blockIdx.y * 128 + wm * 64;
  const int n0   = blockIdx.x * 256 + wn * 64;

  const v8f zc = {0.f, 0.f, 0.f, 0.f, 0.f, 0.f, 0.f, 0.f};
  v8f acc[4][4];
  #pragma unroll
  for (int i = 0; i < 4; ++i)
    #pragma unroll
    for (int j = 0; j < 4; ++j) acc[i][j] = zc;

  for (int k0 = 0; k0 < K; k0 += 32) {
    v16bf a[4], b[4];
    #pragma unroll
    for (int i = 0; i < 4; ++i) {
      const __bf16* ar = A + (size_t)(m0 + 16 * i + l16) * K + k0 + lh * 16;
      a[i] = load_frag2(ar, ar + 8);
      if (k0 + 64 < K) __builtin_prefetch(ar + 64, 0, 1);
    }
    #pragma unroll
    for (int j = 0; j < 4; ++j) {
      const __bf16* br = Bt + (size_t)(n0 + 16 * j + l16) * K + k0 + lh * 16;
      b[j] = load_frag2(br, br + 8);
      if (k0 + 64 < K) __builtin_prefetch(br + 64, 0, 1);
    }
    #pragma unroll
    for (int i = 0; i < 4; ++i)
      #pragma unroll
      for (int j = 0; j < 4; ++j)
        acc[i][j] = __builtin_amdgcn_wmma_f32_16x16x32_bf16(
            false, a[i], false, b[j], (short)0, acc[i][j], false, false);
  }

  #pragma unroll
  for (int i = 0; i < 4; ++i)
    #pragma unroll
    for (int j = 0; j < 4; ++j) {
      const int col = n0 + 16 * j + l16;
      const float bv = bias[col];
      #pragma unroll
      for (int r = 0; r < 8; ++r) {
        const int row = m0 + 16 * i + r + 8 * lh;
        float xv = acc[i][j][r] + bv;
        if (res) xv += res[(size_t)row * N + col];
        if (EPI == 1) xv = gelu_exact(xv);
        if (Cf) Cf[(size_t)row * N + col] = xv;
        if (Cb) Cb[(size_t)row * N + col] = (__bf16)xv;
      }
    }
}

// ---------------------------------------------------------------------------
// pack q,k f32 [b*s, h*hd] -> bf16 [b,h,s,hd]  (coalesced both sides)
// q is pre-scaled by 1/sqrt(hd) so the flash kernel applies no per-score mul.
// ---------------------------------------------------------------------------
__global__ void qk_pack_kernel(const float* __restrict__ q,
                               const float* __restrict__ k,
                               __bf16* __restrict__ qb,
                               __bf16* __restrict__ kb) {
  int i = blockIdx.x * blockDim.x + threadIdx.x;
  if (i >= MM * DDIM) return;
  int row = i >> 10;          // b*S + s
  int col = i & 1023;         // h*64 + d
  int b = row >> 11, s = row & 2047;
  int h = col >> 6, d = col & 63;
  size_t phs = (((size_t)b * HH + h) * SS + s) * HD + d;
  qb[phs] = (__bf16)(q[i] * 0.125f);   // 1/sqrt(64)
  kb[phs] = (__bf16)k[i];
}

// ---------------------------------------------------------------------------
// v f32 [b, s, h*hd] -> vT bf16 [b, h, hd, s]   (LDS-tiled 32x32 transpose)
// ---------------------------------------------------------------------------
__global__ void vt_pack_kernel(const float* __restrict__ v,
                               __bf16* __restrict__ vtb) {
  __shared__ float t[32][33];
  const int s0 = blockIdx.x * 32;
  const int c0 = blockIdx.y * 32;   // stays within one head (32 | 64)
  const int b  = blockIdx.z;
  const int tx = threadIdx.x, ty = threadIdx.y;  // 32 x 8
  #pragma unroll
  for (int i = 0; i < 32; i += 8)
    t[ty + i][tx] = v[((size_t)(b * SS + s0 + ty + i)) * DDIM + c0 + tx];
  __syncthreads();
  #pragma unroll
  for (int i = 0; i < 32; i += 8) {
    const int col = c0 + ty + i;
    const int h = col >> 6, d = col & 63;
    vtb[(((size_t)b * HH + h) * HD + d) * SS + s0 + tx] = (__bf16)t[tx][ty + i];
  }
}

// ---------------------------------------------------------------------------
// Flash attention: per wave a 32-row q tile; 32 kv per iteration.
// Row softmax stats are computed lane-locally after a C->A relayout through
// LDS (lane = row), so only 1 shfl_xor(16) per reduction + per-row broadcasts.
// qb,kb: [b,h,s,hd] bf16 (q pre-scaled); vtb: [b,h,hd,s]; ctxb: [b,s,h*hd]
// ---------------------------------------------------------------------------
__global__ __launch_bounds__(128)
void flash_attn_kernel(const __bf16* __restrict__ qb,
                       const __bf16* __restrict__ kb,
                       const __bf16* __restrict__ vtb,
                       const int* __restrict__ amask,
                       __bf16* __restrict__ ctxb) {
  const int lane = threadIdx.x & 31;
  const int wave = threadIdx.x >> 5;
  const int lh   = lane >> 4;
  const int l16  = lane & 15;
  const int bh   = blockIdx.y;
  const int b    = bh >> 4;
  const int h    = bh & 15;
  const int q0   = (blockIdx.x * 4 + wave) * 32;

  const __bf16* qbase = qb + ((size_t)bh * SS + q0) * HD;
  const __bf16* kbase = kb + (size_t)bh * SS * HD;
  const __bf16* vbase = vtb + (size_t)bh * HD * SS;

  // q A-fragments: [row-tile i][d-chunk f]
  v16bf qa[2][2];
  #pragma unroll
  for (int i = 0; i < 2; ++i) {
    const __bf16* r = qbase + (size_t)(16 * i + l16) * HD;
    qa[i][0] = load_frag2(r + lh * 16, r + lh * 16 + 8);
    qa[i][1] = load_frag2(r + 32 + lh * 16, r + 32 + lh * 16 + 8);
  }

  const v8f zc = {0.f, 0.f, 0.f, 0.f, 0.f, 0.f, 0.f, 0.f};
  v8f acc[2][4];
  #pragma unroll
  for (int i = 0; i < 2; ++i)
    #pragma unroll
    for (int j = 0; j < 4; ++j) acc[i][j] = zc;
  float mrow[2] = {-1e30f, -1e30f};
  float lrow[2] = {0.f, 0.f};

  __shared__ float sf[4][32][32];  // per-wave score tile, f32, 16 KB

  for (int kv0 = 0; kv0 < SS; kv0 += 32) {
    // ---- score WMMAs ----
    const __bf16* kr0 = kbase + (size_t)(kv0 + l16) * HD;
    const __bf16* kr1 = kbase + (size_t)(kv0 + 16 + l16) * HD;
    v16bf k00 = load_frag2(kr0 + lh * 16, kr0 + lh * 16 + 8);
    v16bf k01 = load_frag2(kr0 + 32 + lh * 16, kr0 + 32 + lh * 16 + 8);
    v16bf k10 = load_frag2(kr1 + lh * 16, kr1 + lh * 16 + 8);
    v16bf k11 = load_frag2(kr1 + 32 + lh * 16, kr1 + 32 + lh * 16 + 8);

    const int mv0 = amask[b * SS + kv0 + l16];
    const int mv1 = amask[b * SS + kv0 + 16 + l16];

    #pragma unroll
    for (int i = 0; i < 2; ++i) {
      v8f s0 = zc, s1 = zc;
      s0 = __builtin_amdgcn_wmma_f32_16x16x32_bf16(false, qa[i][0], false, k00,
                                                   (short)0, s0, false, false);
      s0 = __builtin_amdgcn_wmma_f32_16x16x32_bf16(false, qa[i][1], false, k01,
                                                   (short)0, s0, false, false);
      s1 = __builtin_amdgcn_wmma_f32_16x16x32_bf16(false, qa[i][0], false, k10,
                                                   (short)0, s1, false, false);
      s1 = __builtin_amdgcn_wmma_f32_16x16x32_bf16(false, qa[i][1], false, k11,
                                                   (short)0, s1, false, false);
      // mask, store to LDS in C-layout (f32); scale already folded into q
      #pragma unroll
      for (int r = 0; r < 8; ++r) {
        sf[wave][16 * i + r + 8 * lh][l16]      = mv0 ? s0[r] : -1e9f;
        sf[wave][16 * i + r + 8 * lh][16 + l16] = mv1 ? s1[r] : -1e9f;
      }
    }

    // ---- per-row softmax in A-layout (lane = row), build P fragments ----
    v16bf pa[2];
    float alrow[2];
    #pragma unroll
    for (int i = 0; i < 2; ++i) {
      const float* rowp = &sf[wave][16 * i + l16][0];
      float4 x0 = *(const float4*)(rowp + lh * 8);
      float4 x1 = *(const float4*)(rowp + lh * 8 + 4);
      float4 x2 = *(const float4*)(rowp + 16 + lh * 8);
      float4 x3 = *(const float4*)(rowp + 16 + lh * 8 + 4);
      float sv[16] = {x0.x, x0.y, x0.z, x0.w, x1.x, x1.y, x1.z, x1.w,
                      x2.x, x2.y, x2.z, x2.w, x3.x, x3.y, x3.z, x3.w};
      float t = sv[0];
      #pragma unroll
      for (int e = 1; e < 16; ++e) t = fmaxf(t, sv[e]);
      t = fmaxf(t, __shfl_xor(t, 16));
      const float mn = fmaxf(mrow[i], t);
      alrow[i] = __expf(mrow[i] - mn);
      mrow[i] = mn;
      float sum = 0.f;
      BF16Frag pf;
      #pragma unroll
      for (int e = 0; e < 16; ++e) {
        float p = __expf(sv[e] - mn);
        sum += p;
        pf.v[e] = (__bf16)p;
      }
      sum += __shfl_xor(sum, 16);
      lrow[i] = lrow[i] * alrow[i] + sum;
      pa[i] = pf.v;
    }

    // ---- rescale accumulators (broadcast alpha to C-layout rows) ----
    #pragma unroll
    for (int i = 0; i < 2; ++i)
      #pragma unroll
      for (int r = 0; r < 8; ++r) {
        const float alc = __shfl(alrow[i], r + 8 * lh);
        #pragma unroll
        for (int j = 0; j < 4; ++j) acc[i][j][r] *= alc;
      }

    // ---- ctx WMMAs: acc += P * V ----
    #pragma unroll
    for (int j = 0; j < 4; ++j) {
      const __bf16* vr = vbase + (size_t)(16 * j + l16) * SS + kv0 + lh * 8;
      v16bf vf = load_frag2(vr, vr + 16);
      #pragma unroll
      for (int i = 0; i < 2; ++i)
        acc[i][j] = __builtin_amdgcn_wmma_f32_16x16x32_bf16(
            false, pa[i], false, vf, (short)0, acc[i][j], false, false);
    }
  }

  // ---- normalize + write ctx bf16 in [b, s, h*hd] layout ----
  #pragma unroll
  for (int i = 0; i < 2; ++i) {
    const float linv = 1.0f / lrow[i];
    #pragma unroll
    for (int r = 0; r < 8; ++r) {
      const float lc = __shfl(linv, r + 8 * lh);
      const int row = q0 + 16 * i + r + 8 * lh;
      const size_t o = ((size_t)(b * SS + row)) * DDIM + h * HD;
      #pragma unroll
      for (int j = 0; j < 4; ++j)
        ctxb[o + 16 * j + l16] = (__bf16)(acc[i][j][r] * lc);
    }
  }
}

// ---------------------------------------------------------------------------
// LayerNorm over last dim (D=1024): y = (a(+res) - mu)/sqrt(var+eps)*g + b
// ---------------------------------------------------------------------------
__global__ __launch_bounds__(256)
void ln_kernel(const float* __restrict__ a, const float* __restrict__ res,
               const float* __restrict__ g, const float* __restrict__ bt,
               float eps, float* __restrict__ of, __bf16* __restrict__ ob,
               int D) {
  const int row = blockIdx.x;
  const int tid = threadIdx.x;
  float v[4];
  float s = 0.f, s2 = 0.f;
  #pragma unroll
  for (int i = 0; i < 4; ++i) {
    int c = tid + i * 256;
    float xv = a[(size_t)row * D + c];
    if (res) xv += res[(size_t)row * D + c];
    v[i] = xv;
    s += xv;
    s2 += xv * xv;
  }
  __shared__ float rs[256], rs2[256];
  rs[tid] = s; rs2[tid] = s2;
  __syncthreads();
  for (int off = 128; off > 0; off >>= 1) {
    if (tid < off) { rs[tid] += rs[tid + off]; rs2[tid] += rs2[tid + off]; }
    __syncthreads();
  }
  const float mean = rs[0] / D;
  const float var  = rs2[0] / D - mean * mean;
  const float inv  = rsqrtf(var + eps);
  #pragma unroll
  for (int i = 0; i < 4; ++i) {
    int c = tid + i * 256;
    float y = (v[i] - mean) * inv * g[c] + bt[c];
    if (of) of[(size_t)row * D + c] = y;
    if (ob) ob[(size_t)row * D + c] = (__bf16)y;
  }
}

// ---------------------------------------------------------------------------
extern "C" void kernel_launch(void* const* d_in, const int* in_sizes, int n_in,
                              void* d_out, int out_size, void* d_ws,
                              size_t ws_size, hipStream_t stream) {
  const float* x     = (const float*)d_in[0];
  const int*   amask = (const int*)d_in[1];
  const float* wq = (const float*)d_in[2];
  const float* bq = (const float*)d_in[3];
  const float* wk = (const float*)d_in[4];
  const float* bk = (const float*)d_in[5];
  const float* wv = (const float*)d_in[6];
  const float* bv = (const float*)d_in[7];
  const float* wo = (const float*)d_in[8];
  const float* bo = (const float*)d_in[9];
  const float* ln1_g = (const float*)d_in[10];
  const float* ln1_b = (const float*)d_in[11];
  const float* w1 = (const float*)d_in[12];
  const float* b1 = (const float*)d_in[13];
  const float* w2 = (const float*)d_in[14];
  const float* b2 = (const float*)d_in[15];
  const float* lnf_g = (const float*)d_in[16];
  const float* lnf_b = (const float*)d_in[17];
  const float* ln2_g = (const float*)d_in[18];
  const float* ln2_b = (const float*)d_in[19];
  float* out = (float*)d_out;

  char* base = (char*)d_ws;
  size_t off = 0;
  auto alloc = [&](size_t bytes) -> char* {
    char* p = base + off;
    off += (bytes + 255) & ~(size_t)255;
    return p;
  };
  const size_t MD = (size_t)MM * DDIM;
  __bf16* xb  = (__bf16*)alloc(MD * 2);
  __bf16* wqT = (__bf16*)alloc((size_t)DDIM * DDIM * 2);
  __bf16* wkT = (__bf16*)alloc((size_t)DDIM * DDIM * 2);
  __bf16* wvT = (__bf16*)alloc((size_t)DDIM * DDIM * 2);
  __bf16* woT = (__bf16*)alloc((size_t)DDIM * DDIM * 2);
  __bf16* w1T = (__bf16*)alloc((size_t)FF * DDIM * 2);
  __bf16* w2T = (__bf16*)alloc((size_t)DDIM * FF * 2);
  float*  qf  = (float*)alloc(MD * 4);   // later: pre-LN1, then ff
  float*  kf  = (float*)alloc(MD * 4);   // later: x1
  float*  vf  = (float*)alloc(MD * 4);   // later: ffn2 pre-LN
  __bf16* qb  = (__bf16*)alloc(MD * 2);
  __bf16* kb  = (__bf16*)alloc(MD * 2);
  __bf16* vtb = (__bf16*)alloc(MD * 2);
  __bf16* ctxb = (__bf16*)alloc(MD * 2);
  __bf16* x1b  = (__bf16*)alloc(MD * 2);
  __bf16* hb   = (__bf16*)alloc((size_t)MM * FF * 2);
  float* pre1 = qf;   // x + attn_out
  float* x1   = kf;
  float* ffp  = vf;   // h@w2+b2+x1 before LN
  float* ff   = qf;   // LN(ffp)

  const dim3 tb(32, 8);
  // prep: casts + weight transposes
  cast_bf16_kernel<<<(int)(MD / 256), 256, 0, stream>>>(x, xb, (int)MD);
  transpose_cast_kernel<<<dim3(DDIM / 32, DDIM / 32), tb, 0, stream>>>(wq, wqT, DDIM, DDIM);
  transpose_cast_kernel<<<dim3(DDIM / 32, DDIM / 32), tb, 0, stream>>>(wk, wkT, DDIM, DDIM);
  transpose_cast_kernel<<<dim3(DDIM / 32, DDIM / 32), tb, 0, stream>>>(wv, wvT, DDIM, DDIM);
  transpose_cast_kernel<<<dim3(DDIM / 32, DDIM / 32), tb, 0, stream>>>(wo, woT, DDIM, DDIM);
  transpose_cast_kernel<<<dim3(FF / 32, DDIM / 32), tb, 0, stream>>>(w1, w1T, DDIM, FF);
  transpose_cast_kernel<<<dim3(DDIM / 32, FF / 32), tb, 0, stream>>>(w2, w2T, FF, DDIM);

  // QKV projections
  gemm_kernel<0><<<dim3(DDIM / 256, MM / 128), 256, 0, stream>>>(
      xb, wqT, bq, nullptr, qf, nullptr, MM, DDIM, DDIM);
  gemm_kernel<0><<<dim3(DDIM / 256, MM / 128), 256, 0, stream>>>(
      xb, wkT, bk, nullptr, kf, nullptr, MM, DDIM, DDIM);
  gemm_kernel<0><<<dim3(DDIM / 256, MM / 128), 256, 0, stream>>>(
      xb, wvT, bv, nullptr, vf, nullptr, MM, DDIM, DDIM);
  qk_pack_kernel<<<(int)(MD / 256), 256, 0, stream>>>(qf, kf, qb, kb);
  vt_pack_kernel<<<dim3(SS / 32, DDIM / 32, BB), tb, 0, stream>>>(vf, vtb);

  // flash attention
  flash_attn_kernel<<<dim3(SS / 128, BB * HH), 128, 0, stream>>>(qb, kb, vtb,
                                                                 amask, ctxb);
  // out proj + residual, LN1
  gemm_kernel<0><<<dim3(DDIM / 256, MM / 128), 256, 0, stream>>>(
      ctxb, woT, bo, x, pre1, nullptr, MM, DDIM, DDIM);
  ln_kernel<<<MM, 256, 0, stream>>>(pre1, nullptr, ln1_g, ln1_b, 1e-5f, x1,
                                    x1b, DDIM);
  // FFN
  gemm_kernel<1><<<dim3(FF / 256, MM / 128), 256, 0, stream>>>(
      x1b, w1T, b1, nullptr, nullptr, hb, MM, FF, DDIM);
  gemm_kernel<0><<<dim3(DDIM / 256, MM / 128), 256, 0, stream>>>(
      hb, w2T, b2, x1, ffp, nullptr, MM, DDIM, FF);
  ln_kernel<<<MM, 256, 0, stream>>>(ffp, nullptr, lnf_g, lnf_b, 1e-12f, ff,
                                    nullptr, DDIM);
  // final residual LN -> output
  ln_kernel<<<MM, 256, 0, stream>>>(x1, ff, ln2_g, ln2_b, 1e-5f, out, nullptr,
                                    DDIM);
}
